// MultiHeadAttention_8529805050400
// MI455X (gfx1250) — matrix operand order repaired and checked by reading, not compiled, not run on hardware
//
#include <hip/hip_runtime.h>
#include <hip/hip_bf16.h>

#define B_ 4
#define S_ 2048
#define D_ 1024
#define H_ 16
#define DEPTH_ 64

typedef __attribute__((ext_vector_type(16))) __bf16 bf16x16;
typedef __attribute__((ext_vector_type(8)))  float  f32x8;
typedef __attribute__((ext_vector_type(4)))  int    i32x4;

union Frag {
    unsigned int u[8];
    bf16x16 v;
};

__device__ __forceinline__ f32x8 wmma_bf16(const Frag& a, const Frag& b, f32x8 c) {
    return __builtin_amdgcn_wmma_f32_16x16x32_bf16(false, a.v, false, b.v,
                                                   (short)0, c, false, false);
}

__device__ __forceinline__ unsigned int pack2bf(float x, float y) {
    unsigned int a = __float_as_uint(x);
    unsigned int b = __float_as_uint(y);
    a = (a + 0x7FFFu + ((a >> 16) & 1u)) >> 16;
    b = (b + 0x7FFFu + ((b >> 16) & 1u)) >> 16;
    return (a & 0xFFFFu) | (b << 16);
}

__device__ __forceinline__ unsigned short f2bf(float x) {
    unsigned int a = __float_as_uint(x);
    return (unsigned short)((a + 0x7FFFu + ((a >> 16) & 1u)) >> 16);
}

// ---- optional gfx1250 async global->LDS staging (ASYNCcnt path) ----------
#if defined(__AMDGCN__) && __has_builtin(__builtin_amdgcn_global_load_async_to_lds_b128) && __has_builtin(__builtin_amdgcn_s_wait_asynccnt)
#define HAVE_ASYNC_LDS 1
typedef __attribute__((address_space(1))) i32x4 gas_v4i;
typedef __attribute__((address_space(3))) i32x4 las_v4i;
__device__ __forceinline__ void async_copy16(const void* g, void* l) {
    // flat LDS pointers carry the DS byte offset in addr[31:0] (ISA 10.2);
    // integer round-trips avoid addrspace-cast legality issues.
    __builtin_amdgcn_global_load_async_to_lds_b128(
        (gas_v4i*)(unsigned long long)(size_t)g,
        (las_v4i*)(unsigned int)(size_t)l, 0, 0);
}
__device__ __forceinline__ void async_wait0() {
    __builtin_amdgcn_s_wait_asynccnt(0);
}
#else
#define HAVE_ASYNC_LDS 0
#endif

// --------------------------------------------------------------------------
// Tiled GEMM with bias:  Y(M,N) = X(M,K) @ W(N,K)^T + bias(N)
// 256 threads (8 waves). Block tile 128x128, K-tile 64 (2 WMMA K-steps),
// register-prefetch double buffering (next tile's global loads overlap the
// current tile's 16 WMMAs). Wave tile 32x64 -> 2x4 accumulators of 16x16.
// --------------------------------------------------------------------------
template <bool IN_BF16, bool OUT_F32>
__global__ __launch_bounds__(256) void gemm_bias_kernel(
    const void* __restrict__ Xv, const float* __restrict__ W,
    const float* __restrict__ bias, void* __restrict__ Yv,
    int M, int N, int K)
{
    __shared__ unsigned int lds_a[128][33]; // 128 rows x 64 bf16 (32 dwords) + pad
    __shared__ unsigned int lds_b[128][33]; // 128 cols(N) x 64 bf16 over K

    const int m0   = blockIdx.y * 128;
    const int n0   = blockIdx.x * 128;
    const int t    = threadIdx.x;
    const int lrow = t >> 1;
    const int half = t & 1;
    const int lane = t & 31;
    const int wave = t >> 5;
    const int wm   = wave & 3;  // 4 M-groups of 32 rows
    const int wn   = wave >> 2; // 2 N-groups of 64 cols
    const int g    = lane >> 4;
    const int nm   = lane & 15;

    f32x8 acc[2][4];
#pragma unroll
    for (int i = 0; i < 2; ++i)
#pragma unroll
        for (int j = 0; j < 4; ++j)
            acc[i][j] = {};

    unsigned int pa[16], pb[16];

    auto load_a = [&](int kt) {
        if (IN_BF16) {
            const unsigned short* X = (const unsigned short*)Xv;
            const uint4* src = reinterpret_cast<const uint4*>(
                X + (size_t)(m0 + lrow) * K + kt * 64 + half * 32);
#pragma unroll
            for (int i = 0; i < 4; ++i) {
                uint4 u = src[i];
                pa[4 * i + 0] = u.x; pa[4 * i + 1] = u.y;
                pa[4 * i + 2] = u.z; pa[4 * i + 3] = u.w;
            }
        } else {
            const float* X = (const float*)Xv;
            const float4* src = reinterpret_cast<const float4*>(
                X + (size_t)(m0 + lrow) * K + kt * 64 + half * 32);
#pragma unroll
            for (int i = 0; i < 8; ++i) {
                float4 f = src[i];
                pa[2 * i + 0] = pack2bf(f.x, f.y);
                pa[2 * i + 1] = pack2bf(f.z, f.w);
            }
        }
    };
    auto load_b = [&](int kt) {
        const float4* src = reinterpret_cast<const float4*>(
            W + (size_t)(n0 + lrow) * K + kt * 64 + half * 32);
#pragma unroll
        for (int i = 0; i < 8; ++i) {
            float4 f = src[i];
            pb[2 * i + 0] = pack2bf(f.x, f.y);
            pb[2 * i + 1] = pack2bf(f.z, f.w);
        }
    };

    const int nkt = K >> 6;
    load_a(0);
    load_b(0);

    for (int kt = 0; kt < nkt; ++kt) {
        __syncthreads(); // previous tile's LDS reads complete
#pragma unroll
        for (int i = 0; i < 16; ++i) {
            lds_a[lrow][half * 16 + i] = pa[i];
            lds_b[lrow][half * 16 + i] = pb[i];
        }
        __syncthreads();

        if (kt + 1 < nkt) { // prefetch next tile, overlaps WMMAs below
            load_a(kt + 1);
            load_b(kt + 1);
        }

#pragma unroll
        for (int ks = 0; ks < 2; ++ks) {
            Frag af[2], bfg[4];
#pragma unroll
            for (int i = 0; i < 2; ++i) {
                int row = wm * 32 + i * 16 + nm;
#pragma unroll
                for (int vv = 0; vv < 4; ++vv) af[i].u[vv] = lds_a[row][ks * 16 + g * 4 + vv];
#pragma unroll
                for (int vv = 4; vv < 8; ++vv) af[i].u[vv] = lds_a[row][ks * 16 + g * 4 + 4 + vv];
            }
#pragma unroll
            for (int j = 0; j < 4; ++j) {
                int col = wn * 64 + j * 16 + nm;
#pragma unroll
                for (int vv = 0; vv < 8; ++vv) bfg[j].u[vv] = lds_b[col][ks * 16 + g * 4 + vv + (vv >= 4 ? 4 : 0)];
            }
#pragma unroll
            for (int i = 0; i < 2; ++i)
#pragma unroll
                for (int j = 0; j < 4; ++j)
                    acc[i][j] = wmma_bf16(af[i], bfg[j], acc[i][j]);
        }
    }

    // ---- epilogue: bias add + store ----
#pragma unroll
    for (int i = 0; i < 2; ++i) {
#pragma unroll
        for (int j = 0; j < 4; ++j) {
            int col = n0 + wn * 64 + j * 16 + nm;
            float bv = bias[col];
#pragma unroll
            for (int vv = 0; vv < 8; ++vv) {
                int row = m0 + wm * 32 + i * 16 + vv + 8 * g;
                float val = acc[i][j][vv] + bv;
                if (OUT_F32)
                    ((float*)Yv)[(size_t)row * N + col] = val;
                else
                    ((unsigned short*)Yv)[(size_t)row * N + col] = f2bf(val);
            }
        }
    }
}

// --------------------------------------------------------------------------
// Flash attention (causal) on bf16 projections, fp32 online softmax.
// Grid: (B*H, S/64). Block: 128 threads = 4 waves; wave w owns q rows
// [qb*64 + w*16, +16). Key tiles of 64; depth = 64 = 2 WMMA K-steps.
// Q/K tiles staged with async global->LDS b128 when available.
// --------------------------------------------------------------------------
__global__ __launch_bounds__(128) void attn_kernel(
    const unsigned short* __restrict__ Qp,
    const unsigned short* __restrict__ Kp,
    const unsigned short* __restrict__ Vp,
    unsigned short* __restrict__ Op)
{
    // stride 36 dwords = 144B: 16B-aligned rows for async B128, odd-ish mod 64 banks
    __shared__ __attribute__((aligned(16))) unsigned int lds_q[64][36];
    __shared__ __attribute__((aligned(16))) unsigned int lds_k[64][36];
    __shared__ unsigned short lds_v[64][68];     // [depth][key] transposed
    __shared__ unsigned short lds_p[4][16][68];  // per-wave P (16 x 64 keys)

    const int bh = blockIdx.x;
    const int b  = bh / H_;
    const int h  = bh % H_;
    const int qb = blockIdx.y;
    const int q0 = qb * 64;

    const int t    = threadIdx.x;
    const int lrow = t >> 1;
    const int half = t & 1;
    const int lane = t & 31;
    const int w    = t >> 5;
    const int g    = lane >> 4;
    const int nm   = lane & 15;

    // ---- stage Q tile once ----
    {
        const uint4* src = reinterpret_cast<const uint4*>(
            Qp + (size_t)(b * S_ + q0 + lrow) * D_ + h * DEPTH_ + half * 32);
#if HAVE_ASYNC_LDS
#pragma unroll
        for (int i = 0; i < 4; ++i)
            async_copy16(src + i, &lds_q[lrow][half * 16 + 4 * i]);
        async_wait0();
#else
#pragma unroll
        for (int i = 0; i < 4; ++i) {
            uint4 u = src[i];
            lds_q[lrow][half * 16 + 4 * i + 0] = u.x;
            lds_q[lrow][half * 16 + 4 * i + 1] = u.y;
            lds_q[lrow][half * 16 + 4 * i + 2] = u.z;
            lds_q[lrow][half * 16 + 4 * i + 3] = u.w;
        }
#endif
    }
    __syncthreads();

    Frag qf[2];
#pragma unroll
    for (int ks = 0; ks < 2; ++ks) {
        int row = w * 16 + nm;
#pragma unroll
        for (int vv = 0; vv < 4; ++vv) qf[ks].u[vv] = lds_q[row][ks * 16 + g * 4 + vv];
#pragma unroll
        for (int vv = 4; vv < 8; ++vv) qf[ks].u[vv] = lds_q[row][ks * 16 + g * 4 + 4 + vv];
    }

    f32x8 acc_o[4];
#pragma unroll
    for (int t4 = 0; t4 < 4; ++t4) acc_o[t4] = {};
    float row_max[8], row_sum[8];
#pragma unroll
    for (int vv = 0; vv < 8; ++vv) { row_max[vv] = -3.0e38f; row_sum[vv] = 0.0f; }

    for (int kt = 0; kt <= qb; ++kt) {
        // ---- stage K tile (async if available) and transposed V tile ----
        {
            const uint4* ksrc = reinterpret_cast<const uint4*>(
                Kp + (size_t)(b * S_ + kt * 64 + lrow) * D_ + h * DEPTH_ + half * 32);
#if HAVE_ASYNC_LDS
#pragma unroll
            for (int i = 0; i < 4; ++i)
                async_copy16(ksrc + i, &lds_k[lrow][half * 16 + 4 * i]);
#else
#pragma unroll
            for (int i = 0; i < 4; ++i) {
                uint4 u = ksrc[i];
                lds_k[lrow][half * 16 + 4 * i + 0] = u.x;
                lds_k[lrow][half * 16 + 4 * i + 1] = u.y;
                lds_k[lrow][half * 16 + 4 * i + 2] = u.z;
                lds_k[lrow][half * 16 + 4 * i + 3] = u.w;
            }
#endif
            const uint4* vsrc = reinterpret_cast<const uint4*>(
                Vp + (size_t)(b * S_ + kt * 64 + lrow) * D_ + h * DEPTH_ + half * 32);
#pragma unroll
            for (int i = 0; i < 4; ++i) {
                uint4 u = vsrc[i];
                unsigned int uu[4] = { u.x, u.y, u.z, u.w };
#pragma unroll
                for (int c = 0; c < 4; ++c) {
                    int dep = half * 32 + 8 * i + 2 * c;
                    lds_v[dep + 0][lrow] = (unsigned short)(uu[c] & 0xFFFFu);
                    lds_v[dep + 1][lrow] = (unsigned short)(uu[c] >> 16);
                }
            }
#if HAVE_ASYNC_LDS
            async_wait0();
#endif
        }
        __syncthreads();

        // ---- logits: S = Q @ K^T, 4 key subtiles of 16 ----
        f32x8 sacc[4];
#pragma unroll
        for (int j = 0; j < 4; ++j) {
            f32x8 s = {};
#pragma unroll
            for (int ks = 0; ks < 2; ++ks) {
                Frag kf;
                int key = j * 16 + nm;
#pragma unroll
                for (int vv = 0; vv < 8; ++vv) kf.u[vv] = lds_k[key][ks * 16 + g * 4 + vv + (vv >= 4 ? 4 : 0)];
                s = wmma_bf16(qf[ks], kf, s);
            }
            sacc[j] = s;
        }

        // ---- scale + causal mask (diagonal tile only) ----
#pragma unroll
        for (int j = 0; j < 4; ++j) {
#pragma unroll
            for (int vv = 0; vv < 8; ++vv) {
                float sv = sacc[j][vv] * 0.125f; // 1/sqrt(64)
                if (kt == qb) {
                    int key_g = kt * 64 + j * 16 + nm;
                    int q_g   = q0 + w * 16 + vv + 8 * g;
                    if (key_g > q_g) sv += -1.0e9f;
                }
                sacc[j][vv] = sv;
            }
        }

        // ---- online softmax: row = vv + 8*g, spread over 16 lanes ----
#pragma unroll
        for (int vv = 0; vv < 8; ++vv) {
            float tmax = fmaxf(fmaxf(sacc[0][vv], sacc[1][vv]),
                               fmaxf(sacc[2][vv], sacc[3][vv]));
#pragma unroll
            for (int m = 1; m < 16; m <<= 1) tmax = fmaxf(tmax, __shfl_xor(tmax, m, 32));
            float mnew = fmaxf(row_max[vv], tmax);
            float corr = __expf(row_max[vv] - mnew);
            row_max[vv] = mnew;
            float ls = 0.0f;
#pragma unroll
            for (int j = 0; j < 4; ++j) {
                float p = __expf(sacc[j][vv] - mnew);
                sacc[j][vv] = p;
                ls += p;
            }
#pragma unroll
            for (int m = 1; m < 16; m <<= 1) ls += __shfl_xor(ls, m, 32);
            row_sum[vv] = row_sum[vv] * corr + ls;
#pragma unroll
            for (int t4 = 0; t4 < 4; ++t4) acc_o[t4][vv] *= corr;
        }

        // ---- spill P (C-layout) to per-wave LDS as bf16 ----
#pragma unroll
        for (int j = 0; j < 4; ++j)
#pragma unroll
            for (int vv = 0; vv < 8; ++vv)
                lds_p[w][vv + 8 * g][j * 16 + nm] = f2bf(sacc[j][vv]);

        // ---- O += P @ V (reload P in A-layout; V fragments from transposed LDS) ----
#pragma unroll
        for (int ks = 0; ks < 2; ++ks) {
            Frag pf;
#pragma unroll
            for (int vv = 0; vv < 4; ++vv)
                pf.u[vv] = *(const unsigned int*)&lds_p[w][nm][2 * (ks * 16 + g * 4 + vv)];
#pragma unroll
            for (int vv = 4; vv < 8; ++vv)
                pf.u[vv] = *(const unsigned int*)&lds_p[w][nm][2 * (ks * 16 + g * 4 + 4 + vv)];
#pragma unroll
            for (int t4 = 0; t4 < 4; ++t4) {
                Frag vf;
                int dep = t4 * 16 + nm;
#pragma unroll
                for (int vv = 0; vv < 8; ++vv)
                    vf.u[vv] = *(const unsigned int*)&lds_v[dep][2 * (ks * 16 + g * 4 + vv + (vv >= 4 ? 4 : 0))];
                acc_o[t4] = wmma_bf16(pf, vf, acc_o[t4]);
            }
        }
        __syncthreads();
    }

    // ---- normalize and store O as bf16 ----
#pragma unroll
    for (int t4 = 0; t4 < 4; ++t4) {
#pragma unroll
        for (int vv = 0; vv < 8; ++vv) {
            int row = q0 + w * 16 + vv + 8 * g;
            int dep = t4 * 16 + nm;
            float val = acc_o[t4][vv] / row_sum[vv];
            Op[(size_t)(b * S_ + row) * D_ + h * DEPTH_ + dep] = f2bf(val);
        }
    }
}

// --------------------------------------------------------------------------
extern "C" void kernel_launch(void* const* d_in, const int* in_sizes, int n_in,
                              void* d_out, int out_size, void* d_ws, size_t ws_size,
                              hipStream_t stream) {
    (void)in_sizes; (void)n_in; (void)out_size; (void)ws_size;
    const float* q    = (const float*)d_in[0];
    const float* k    = (const float*)d_in[1];
    const float* v    = (const float*)d_in[2];
    const float* wq_w = (const float*)d_in[3];
    const float* wq_b = (const float*)d_in[4];
    const float* wk_w = (const float*)d_in[5];
    const float* wk_b = (const float*)d_in[6];
    const float* wv_w = (const float*)d_in[7];
    const float* wv_b = (const float*)d_in[8];
    const float* dw   = (const float*)d_in[9];
    const float* db   = (const float*)d_in[10];

    const size_t elems = (size_t)B_ * S_ * D_;
    unsigned short* Qp = (unsigned short*)d_ws;
    unsigned short* Kp = Qp + elems;
    unsigned short* Vp = Kp + elems;
    unsigned short* Op = Vp + elems;

    dim3 gg(D_ / 128, (B_ * S_) / 128);
    gemm_bias_kernel<false, false><<<gg, 256, 0, stream>>>(q, wq_w, wq_b, Qp, B_ * S_, D_, D_);
    gemm_bias_kernel<false, false><<<gg, 256, 0, stream>>>(k, wk_w, wk_b, Kp, B_ * S_, D_, D_);
    gemm_bias_kernel<false, false><<<gg, 256, 0, stream>>>(v, wv_w, wv_b, Vp, B_ * S_, D_, D_);

    attn_kernel<<<dim3(B_ * H_, S_ / 64), 128, 0, stream>>>(Qp, Kp, Vp, Op);

    gemm_bias_kernel<true, true><<<gg, 256, 0, stream>>>(Op, dw, db, (float*)d_out, B_ * S_, D_, D_);
}
